// KNNSegmentator_54735063220388
// MI455X (gfx1250) — matrix operand order repaired and checked by reading, not compiled, not run on hardware
//
#include <hip/hip_runtime.h>
#include <stdint.h>

// ---------------- problem constants ----------------
#define KSEL     20          // top-k
#define NCLS     21          // classes
#define DIMK     384
#define NTRAIN   50000
#define MROWS    1568        // 8 * 196
#define MTILES   98          // 1568 / 16
#define KTILES   12          // 384 / 32
#define NTILES   3125        // 50000 / 16
#define CHUNK_NT 32          // n-tiles per gemm block (512 cols)
#define CHUNK_COLS 512
#define NCHUNKS  98          // ceil(3125/32)
#define NEG_INF  (-3.402823466e38f)

// ---------------- workspace layout (bytes) ----------------
// A packed bf16 frags : 1568*384*2  = 1,204,224
// B packed bf16 frags : 50000*384*2 = 38,400,000
// candV (f32)         : 1568*98*20*4 = 12,293,120
// candI (i32)         : 12,293,120           total ~64.2 MB
#define A_BYTES   1204224u
#define CANDV_OFF 39604224u
#define CANDI_OFF 51897344u

typedef __attribute__((ext_vector_type(16))) __bf16 v16bf;
typedef __attribute__((ext_vector_type(8)))  float  v8f;

union FragU { uint4 u[2]; v16bf b; };

__device__ __forceinline__ unsigned int f2bf(float f) {
  unsigned int u = __float_as_uint(f);
  return (u + 0x7FFFu + ((u >> 16) & 1u)) >> 16;   // RNE f32 -> bf16
}

// ---------------- kernel 1: pack A (test features) into WMMA A-frag layout ----
// A-frag (16x32 bf16, ISA 7.12.2): lane = m%16 + 16*g ; k = 16*(v/4) + 8*g + 2*(v%4) + {0,1}
__global__ __launch_bounds__(256)
void pack_a(const float* __restrict__ A, unsigned int* __restrict__ Ap) {
  int tid = blockIdx.x * 256 + threadIdx.x;          // 1568*192 threads
  int kp = tid % (DIMK / 2);
  int m  = tid / (DIMK / 2);
  float2 t = *(const float2*)(A + (size_t)m * DIMK + 2 * kp);
  unsigned int pk = f2bf(t.x) | (f2bf(t.y) << 16);
  int k0 = 2 * kp;
  int ktile = k0 >> 5, kin = k0 & 31;
  int hi = kin >> 4, rem = kin & 15;
  int g = rem >> 3;
  int v = 4 * hi + ((rem & 7) >> 1);
  int lane = (m & 15) + 16 * g;
  int mtile = m >> 4;
  Ap[((size_t)(mtile * KTILES + ktile) * 32 + lane) * 8 + v] = pk;
}

// ---------------- kernel 2: pack B (train features) into WMMA B-frag layout ----
// B-frag (32x16 bf16, ISA 7.12.4): lane = n%16 + 16*g ; k = 16*(v/4) + 8*g + 2*(v%4) + {0,1}
__global__ __launch_bounds__(256)
void pack_b(const float* __restrict__ B, unsigned int* __restrict__ Bp) {
  int tid = blockIdx.x * 256 + threadIdx.x;          // 50000*192 threads
  int n  = tid % NTRAIN;
  int kp = tid / NTRAIN;
  int k0 = 2 * kp;
  float f0 = B[(size_t)k0 * NTRAIN + n];
  float f1 = B[(size_t)(k0 + 1) * NTRAIN + n];
  unsigned int pk = f2bf(f0) | (f2bf(f1) << 16);
  int ktile = k0 >> 5, kin = k0 & 31;
  int g = (kin >> 3) & 1;
  int v = 4 * (kin >> 4) + ((kin & 7) >> 1);
  int lane = (n & 15) + 16 * g;
  int ntile = n >> 4;
  Bp[((size_t)(ntile * KTILES + ktile) * 32 + lane) * 8 + v] = pk;
}

// load k-step `kt` fragments for all 4 n-tiles into dst (8 x b128 per wave-lane group)
#define LOAD4T(dst, kt)                                                         \
  do {                                                                          \
    _Pragma("unroll")                                                           \
    for (int t = 0; t < 4; ++t) {                                               \
      dst[t].u[0] = bp[t][(kt) * 64];                                           \
      dst[t].u[1] = bp[t][(kt) * 64 + 1];                                       \
    }                                                                           \
  } while (0)

// ---------------- kernel 3: fused bf16-WMMA GEMM + per-chunk exact top-20 ----
// __launch_bounds__(256, 1): relax occupancy so 2 in-flight B buffer groups +
// 12 A fragments + 4 accumulator chains fit in registers (pipelined inner loop).
__global__ __launch_bounds__(256, 1)
void gemm_topk(const unsigned int* __restrict__ Ap, const unsigned int* __restrict__ Bp,
               float* __restrict__ candV, int* __restrict__ candI) {
  __shared__ float sim[16][CHUNK_COLS];              // 32 KB, one 16x512 sim tile
  const int tid = threadIdx.x, w = tid >> 5, lane = tid & 31;
  const uint4* A4 = (const uint4*)Ap;
  const uint4* B4 = (const uint4*)Bp;

  // A fragments for this M-tile resident in registers (reused by all 4 n-tiles)
  FragU afr[KTILES];
  {
    const uint4* ap = A4 + (size_t)blockIdx.x * KTILES * 64 + lane * 2;
    #pragma unroll
    for (int kt = 0; kt < KTILES; ++kt) { afr[kt].u[0] = ap[kt * 64]; afr[kt].u[1] = ap[kt * 64 + 1]; }
  }

  // 4 n-tiles per wave, processed simultaneously: 4 independent WMMA chains,
  // double-buffered B fragment groups (loads for kt+1 overlap WMMAs of kt).
  const int nt0 = blockIdx.y * CHUNK_NT + w * 4;
  const uint4* bp[4];
  bool valid[4];
  #pragma unroll
  for (int t = 0; t < 4; ++t) {
    int nt = nt0 + t;
    valid[t] = nt < NTILES;
    int ntc = valid[t] ? nt : 0;                     // clamp OOB to tile 0 (safe reads)
    bp[t] = B4 + (size_t)ntc * KTILES * 64 + lane * 2;
    __builtin_prefetch(bp[t], 0, 1);                 // global_prefetch_b8 on the B stream
  }

  v8f acc[4];
  #pragma unroll
  for (int t = 0; t < 4; ++t) acc[t] = (v8f){};

  FragU bufA[4], bufB[4];
  LOAD4T(bufA, 0);
  #pragma unroll
  for (int kt = 0; kt < KTILES; ++kt) {
    FragU (&cur)[4] = (kt & 1) ? bufB : bufA;        // kt is constant after unroll
    FragU (&nxt)[4] = (kt & 1) ? bufA : bufB;
    if (kt + 1 < KTILES) LOAD4T(nxt, kt + 1);        // next group in flight
    #pragma unroll
    for (int t = 0; t < 4; ++t)
      acc[t] = __builtin_amdgcn_wmma_f32_16x16x32_bf16(
          false, afr[kt].b, false, cur[t].b, (short)0, acc[t], false, false);
  }

  // C layout: lanes 0-15 -> N=lane, M=v ; lanes 16-31 -> N=lane-16, M=v+8
  {
    int rb = (lane >> 4) * 8;
    #pragma unroll
    for (int t = 0; t < 4; ++t) {
      int col = (w * 4 + t) * 16 + (lane & 15);
      #pragma unroll
      for (int v = 0; v < 8; ++v) sim[rb + v][col] = valid[t] ? acc[t][v] : NEG_INF;
    }
  }
  __syncthreads();

  // ---- exact per-chunk top-20 per row; each wave owns 2 rows ----
  const int nbase = blockIdx.y * CHUNK_COLS;
  for (int rr = 0; rr < 2; ++rr) {
    int r = w * 2 + rr;
    float lv[16]; int li[16];                        // per-lane sorted ascending (min at [0])
    #pragma unroll
    for (int t = 0; t < 16; ++t) { lv[t] = NEG_INF; li[t] = 0; }
    #pragma unroll
    for (int j = 0; j < 16; ++j) {
      int c = lane + j * 32;                         // bank-conflict-free
      float v = sim[r][c];
      int  n = nbase + c;
      if (v > lv[0]) {
        bool placed = false;
        #pragma unroll
        for (int t = 0; t < 15; ++t) {
          if (!placed) {
            if (v > lv[t + 1]) { lv[t] = lv[t + 1]; li[t] = li[t + 1]; }
            else               { lv[t] = v; li[t] = n; placed = true; }
          }
        }
        if (!placed) { lv[15] = v; li[15] = n; }
      }
    }
    size_t base = ((size_t)(blockIdx.x * 16 + r) * NCHUNKS + blockIdx.y) * KSEL;
    for (int kk = 0; kk < KSEL; ++kk) {
      float bv = lv[15]; int bi = li[15]; int bl = lane;
      #pragma unroll
      for (int off = 16; off > 0; off >>= 1) {       // wave32 argmax (tie: smaller n)
        float ov = __shfl_xor(bv, off, 32);
        int   oi = __shfl_xor(bi, off, 32);
        int   ol = __shfl_xor(bl, off, 32);
        if (ov > bv || (ov == bv && oi < bi)) { bv = ov; bi = oi; bl = ol; }
      }
      if (lane == 0) { candV[base + kk] = bv; candI[base + kk] = bi; }
      if (lane == bl) {                              // winner pops its max
        #pragma unroll
        for (int t = 15; t > 0; --t) { lv[t] = lv[t - 1]; li[t] = li[t - 1]; }
        lv[0] = NEG_INF; li[0] = 0;
      }
    }
  }
}

// ---------------- kernel 4: merge candidates -> top-20 -> softmax vote -> argmax ----
__global__ __launch_bounds__(256)
void merge_vote(const float* __restrict__ candV, const int* __restrict__ candI,
                const int* __restrict__ labels, float* __restrict__ out) {
  const int NC = NCHUNKS * KSEL;                     // 1960 candidates per row
  __shared__ float cv[NCHUNKS * KSEL];
  __shared__ int   ci[NCHUNKS * KSEL];
  __shared__ float selV[KSEL];
  __shared__ int   selI[KSEL];
  __shared__ float redV[8];
  __shared__ int   redP[8];
  const int tid = threadIdx.x;
  const int row = blockIdx.x;
  size_t base = (size_t)row * NC;
  for (int t = tid; t < NC; t += 256) { cv[t] = candV[base + t]; ci[t] = candI[base + t]; }
  __syncthreads();

  for (int it = 0; it < KSEL; ++it) {
    float lv = NEG_INF; int lp = 0;
    for (int t = tid; t < NC; t += 256) { float v = cv[t]; if (v > lv) { lv = v; lp = t; } }
    #pragma unroll
    for (int off = 16; off > 0; off >>= 1) {
      float ov = __shfl_xor(lv, off, 32);
      int   op = __shfl_xor(lp, off, 32);
      if (ov > lv || (ov == lv && op < lp)) { lv = ov; lp = op; }
    }
    if ((tid & 31) == 0) { redV[tid >> 5] = lv; redP[tid >> 5] = lp; }
    __syncthreads();
    if (tid == 0) {
      float bv = redV[0]; int bp = redP[0];
      #pragma unroll
      for (int q = 1; q < 8; ++q) if (redV[q] > bv) { bv = redV[q]; bp = redP[q]; }
      selV[it] = bv; selI[it] = ci[bp]; cv[bp] = NEG_INF;
    }
    __syncthreads();
  }

  // softmax weights over the 20 neighbors (temperature = 1)
  float wgt[KSEL];
  float mx = selV[0];
  #pragma unroll
  for (int k = 1; k < KSEL; ++k) mx = fmaxf(mx, selV[k]);
  float s = 0.f;
  #pragma unroll
  for (int k = 0; k < KSEL; ++k) { wgt[k] = __expf(selV[k] - mx); s += wgt[k]; }
  float inv = 1.0f / s;

  // each thread = one pixel l of the 16x16 patch; label gathers hit L2 (51 MB < 192 MB)
  int lbl[KSEL];
  #pragma unroll
  for (int k = 0; k < KSEL; ++k) lbl[k] = labels[(size_t)tid * NTRAIN + selI[k]];
  float best = -1.f; int bc = 0;
  for (int c = 0; c < NCLS; ++c) {
    float a = 0.f;
    #pragma unroll
    for (int k = 0; k < KSEL; ++k) a += (lbl[k] == c) ? wgt[k] : 0.f;
    a *= inv;
    if (a > best) { best = a; bc = c; }              // first max wins, matches argmax
  }
  int b = row / 196, p = row % 196;
  int pr = p / 14, pc = p % 14;
  int ii = tid >> 4, jj = tid & 15;
  out[(size_t)b * 224 * 224 + (size_t)(pr * 16 + ii) * 224 + (pc * 16 + jj)] = (float)bc;
}

// ---------------- launcher ----------------
extern "C" void kernel_launch(void* const* d_in, const int* in_sizes, int n_in,
                              void* d_out, int out_size, void* d_ws, size_t ws_size,
                              hipStream_t stream) {
  const float* test   = (const float*)d_in[0];   // (8,196,384) f32
  const float* train  = (const float*)d_in[1];   // (384,50000) f32
  const int*   labels = (const int*)d_in[2];     // (256,50000) i32
  float* out = (float*)d_out;                    // (8,224,224) as f32

  char* ws = (char*)d_ws;                        // needs ~64.2 MB
  unsigned int* Ap = (unsigned int*)(ws + 0);
  unsigned int* Bp = (unsigned int*)(ws + A_BYTES);
  float* candV     = (float*)(ws + CANDV_OFF);
  int*   candI     = (int*)(ws + CANDI_OFF);

  pack_a<<<(MROWS * (DIMK / 2)) / 256, 256, 0, stream>>>(test, Ap);        // 1176 blocks
  pack_b<<<(NTRAIN * (DIMK / 2)) / 256, 256, 0, stream>>>(train, Bp);      // 37500 blocks
  gemm_topk<<<dim3(MTILES, NCHUNKS), 256, 0, stream>>>(Ap, Bp, candV, candI);
  merge_vote<<<MROWS, 256, 0, stream>>>(candV, candI, labels, out);
}
